// SelfAttLayer_13675175870585
// MI455X (gfx1250) — compile-verified, matrix operand order checked by reading, not run on hardware
//
#include <hip/hip_runtime.h>
#include <hip/hip_bf16.h>

typedef __attribute__((ext_vector_type(16))) __bf16 v16bf;
typedef __attribute__((ext_vector_type(8)))  float  v8f;
typedef unsigned int v4u __attribute__((ext_vector_type(4)));
typedef int v4i __attribute__((ext_vector_type(4)));
typedef int v8i __attribute__((ext_vector_type(8)));

#define A_    2048
#define T_    91
#define D_    256
#define H_    4
#define DH_   64
#define KD_   1024
#define MROWS (A_ * T_)            // 186368 (== 11648 * 16, no partial tiles)
#define MT    ((MROWS + 15) / 16)  // 11648
#define NEGBIG (-1.0e10f)

// ---------------------------------------------------------------------------
// WMMA helpers (CDNA5: D = A(16x32 bf16) x B(32x16 bf16) + C(16x16 f32))
// ---------------------------------------------------------------------------
__device__ __forceinline__ v8f wmma_bf16(v16bf a, v16bf b, v8f c) {
  return __builtin_amdgcn_wmma_f32_16x16x32_bf16(
      /*neg_a=*/false, a, /*neg_b=*/false, b,
      /*c_mod=*/(short)0, c, /*reuse_a=*/false, /*reuse_b=*/false);
}

__device__ __forceinline__ v16bf fzero16() {
  v16bf f;
#pragma unroll
  for (int i = 0; i < 16; ++i) f[i] = (__bf16)0.0f;
  return f;
}

__device__ __forceinline__ v8f azero() {
  v8f c;
#pragma unroll
  for (int i = 0; i < 8; ++i) c[i] = 0.0f;
  return c;
}

// ISA 7.12.2: 16-bit A fragment = per lane two contiguous 8-element chunks at
// K = g*8 and K = 16 + g*8 (g = lane>>4) of row m = lane&15.  B fragments use
// the same pattern when loaded from a transposed (out-major) source.
__device__ __forceinline__ v16bf frag_ld_bf16(const __bf16* p) {
  v16bf f;
#pragma unroll
  for (int i = 0; i < 8; ++i) { f[i] = p[i]; f[i + 8] = p[16 + i]; }
  return f;
}

__device__ __forceinline__ v16bf frag_ld_f32(const float* p) {
  v16bf f;
#pragma unroll
  for (int i = 0; i < 8; ++i) { f[i] = (__bf16)p[i]; f[i + 8] = (__bf16)p[16 + i]; }
  return f;
}

// ---------------------------------------------------------------------------
// TDM: 1-row (contiguous) tensor_load_to_lds of n bf16 elements.
// D# per CDNA5 ISA.  Issued once per wave
// (EXEC ignored by TDM); completion via s_wait_tensorcnt.
// ---------------------------------------------------------------------------
__device__ __forceinline__ void tdm_load_row_bf16(unsigned lds_off,
                                                  const void* gptr, unsigned n) {
  unsigned long long ga = (unsigned long long)gptr;
  v4u g0;
  g0[0] = 1u;                                            // count=1, user D#
  g0[1] = lds_off;                                       // lds_addr (bytes)
  g0[2] = (unsigned)(ga & 0xFFFFFFFFu);                  // global_addr[31:0]
  g0[3] = (unsigned)((ga >> 32) & 0x01FFFFFFu) | (2u << 30);  // addr[56:32], type=2
  v8i g1;
  g1[0] = 0x00010000;                                    // data_size=1 (2B)
  g1[1] = (int)((n & 0xFFFFu) << 16);                    // tensor_dim0[15:0]
  g1[2] = (int)(((n >> 16) & 0xFFFFu) | (1u << 16));     // tdim0[31:16], tdim1=1
  g1[3] = (int)((n & 0xFFFFu) << 16);                    // tile_dim0 = n
  g1[4] = 1;                                             // tile_dim1 = 1
  g1[5] = (int)n;                                        // tensor_dim0_stride
  g1[6] = 0;
  g1[7] = 0;
  v4i z4 = {0, 0, 0, 0};
#if defined(__clang_major__) && (__clang_major__ >= 23)
  v8i z8 = {0, 0, 0, 0, 0, 0, 0, 0};
  __builtin_amdgcn_tensor_load_to_lds(g0, g1, z4, z4, z8, 0);
#else
  __builtin_amdgcn_tensor_load_to_lds(g0, g1, z4, z4, 0);
#endif
}

// ---------------------------------------------------------------------------
// K0: fp32 -> bf16 weight transpose  dst[o*rows + i] = src[i*cols + o]
// ---------------------------------------------------------------------------
__global__ void transpose_bf16(__bf16* __restrict__ dst, const float* __restrict__ src,
                               int rows, int cols) {
  long long i = (long long)blockIdx.x * blockDim.x + threadIdx.x;
  long long total = (long long)rows * cols;
  if (i < total) {
    int r = (int)(i / cols), c = (int)(i % cols);
    dst[(size_t)c * rows + r] = (__bf16)src[(size_t)r * cols + c];
  }
}

// ---------------------------------------------------------------------------
// K1: fused QKV projection.  grid = MT, block = 384 (12 waves = 3 proj x 4
// heads).  The 16x256 x-tile is staged to LDS as bf16 once (x read 1x from
// HBM, converted once).  Writes relu(x@W+b)(*scale_q) as fp32 into d_out at
// the Qp/Kp/Vp [A,H,T,DH] slots.
// ---------------------------------------------------------------------------
__global__ __launch_bounds__(384)
void qkv_proj(const float* __restrict__ x,
              const __bf16* __restrict__ wtK, const __bf16* __restrict__ wtV,
              const __bf16* __restrict__ wtQ,
              const float* __restrict__ bK, const float* __restrict__ bV,
              const float* __restrict__ bQ, const float* __restrict__ scale_q,
              float* __restrict__ out) {
  const int rt   = blockIdx.x;
  const int tid  = threadIdx.x, lane = tid & 31, w = tid >> 5;  // wave 0..11
  const int nl   = lane & 15, g = lane >> 4;
  const int proj = w >> 2;                 // 0=K, 1=V, 2=Q
  const int hd   = w & 3;                  // head

  __shared__ __bf16 xTile[16][256];        // 8 KB
  for (int i = tid; i < 16 * 256; i += 384) {
    int r0 = i >> 8, c = i & 255;
    int rr = rt * 16 + r0;
    xTile[r0][c] = (rr < MROWS) ? (__bf16)x[(size_t)rr * D_ + c] : (__bf16)0.0f;
  }
  __syncthreads();

  const __bf16* wt   = (proj == 0) ? wtK : (proj == 1) ? wtV : wtQ;
  const float*  bias = (proj == 0) ? bK  : (proj == 1) ? bV  : bQ;
  const size_t  obase = (size_t)((proj == 0) ? 2 : (proj == 1) ? 3 : 1) * (size_t)MROWS * D_;

  v16bf af[8];
#pragma unroll
  for (int c = 0; c < 8; ++c) af[c] = frag_ld_bf16(&xTile[nl][c * 32 + g * 8]);

  v8f acc[4];
#pragma unroll
  for (int j = 0; j < 4; ++j) acc[j] = azero();
#pragma unroll
  for (int kc = 0; kc < 8; ++kc) {                    // K = 256 in steps of 32
#pragma unroll
    for (int j = 0; j < 4; ++j) {
      int col = hd * 64 + j * 16 + nl;                // B column = output index
      v16bf bf = frag_ld_bf16(wt + (size_t)col * D_ + kc * 32 + g * 8);
      acc[j] = wmma_bf16(af[kc], bf, acc[j]);
    }
  }

#pragma unroll
  for (int j = 0; j < 4; ++j) {
    int   col = hd * 64 + j * 16 + nl;
    float bv  = bias[col];
    float sq  = (proj == 2) ? scale_q[col & 63] : 1.0f;
#pragma unroll
    for (int r = 0; r < 8; ++r) {
      int rr = rt * 16 + g * 8 + r;                   // C row m = g*8 + r
      if (rr < MROWS) {
        float v = acc[j][r] + bv;
        v = v > 0.0f ? v : 0.0f;
        v *= sq;
        int a = rr / T_, t = rr % T_;
        out[obase + ((size_t)(a * H_ + hd) * T_ + t) * DH_ + (col & 63)] = v;
      }
    }
  }
}

// ---------------------------------------------------------------------------
// K2: attention.  grid = (A, H), block = 192 (6 waves, one 16-query-row tile
// each; T=91 padded to 96).  K and V^T staged once to LDS as bf16 (K would
// otherwise be read 6x).  Softmax fully in registers via half-wave
// __shfl_xor reductions.  Pad cols get -inf (softmax over 91 like the
// reference); invalid tokens get -1e10.
// ---------------------------------------------------------------------------
__global__ __launch_bounds__(192)
void attn_kernel(const float* __restrict__ out, const unsigned char* __restrict__ pad,
                 __bf16* __restrict__ y1) {
  const int a = blockIdx.x, h = blockIdx.y;
  const int tid = threadIdx.x, lane = tid & 31, w = tid >> 5;
  const int nl = lane & 15, g = lane >> 4;

  __shared__ __bf16 attnLDS[6][16][96];   // 18 KB
  __shared__ __bf16 kLDS[96][64];         // 12 KB (K row-major, zero-padded)
  __shared__ __bf16 vtLDS[64][96];        // 12 KB (V transposed: [dh][t])
  __shared__ float  padLDS[96];

  const size_t ATD = (size_t)MROWS * D_;
  const float* Qb = out + ATD     + ((size_t)(a * H_ + h) * T_) * DH_;
  const float* Kb = out + 2 * ATD + ((size_t)(a * H_ + h) * T_) * DH_;
  const float* Vb = out + 3 * ATD + ((size_t)(a * H_ + h) * T_) * DH_;

  for (int t = tid; t < 96; t += 192)
    padLDS[t] = (t < T_ && pad[(size_t)a * T_ + t]) ? 1.0f : 0.0f;
  for (int i = tid; i < 96 * 64; i += 192) {          // K (contiguous reads)
    int t = i >> 6, d = i & 63;
    kLDS[t][d] = (t < T_) ? (__bf16)Kb[(size_t)t * DH_ + d] : (__bf16)0.0f;
  }
  for (int i = tid; i < 64 * 96; i += 192) {          // V^T
    int d = i / 96, k = i % 96;
    vtLDS[d][k] = (k < T_) ? (__bf16)Vb[(size_t)k * DH_ + d] : (__bf16)0.0f;
  }
  __syncthreads();

  // --- energy = Q K^T (12 WMMAs) -------------------------------------------
  const int  tq  = w * 16 + nl;
  const bool qv  = tq < T_;
  v16bf qf[2];
#pragma unroll
  for (int c = 0; c < 2; ++c)
    qf[c] = qv ? frag_ld_f32(Qb + (size_t)tq * DH_ + c * 32 + g * 8) : fzero16();

  v8f acc[6];
#pragma unroll
  for (int j = 0; j < 6; ++j) {
    acc[j] = azero();
#pragma unroll
    for (int c = 0; c < 2; ++c) {
      v16bf kf = frag_ld_bf16(&kLDS[j * 16 + nl][c * 32 + g * 8]);
      acc[j] = wmma_bf16(qf[c], kf, acc[j]);
    }
  }

  // --- mask + scale (1/sqrt(H) = 0.5) + in-register softmax ----------------
  float padq[8], rmax[8], rsum[8];
#pragma unroll
  for (int r = 0; r < 8; ++r) {
    int t = w * 16 + g * 8 + r;
    padq[r] = (t < T_) ? padLDS[t] : 0.0f;
    rmax[r] = -3.0e38f;
    rsum[r] = 0.0f;
  }
#pragma unroll
  for (int j = 0; j < 6; ++j) {
    int   tk   = j * 16 + nl;
    bool  kin  = tk < T_;
    float padk = kin ? padLDS[tk] : 0.0f;
#pragma unroll
    for (int r = 0; r < 8; ++r) {
      float e;
      if (!kin)                         e = -__builtin_inff();  // pad col
      else if (padq[r] * padk == 0.0f)  e = NEGBIG;             // invalid token
      else                              e = acc[j][r] * 0.5f;
      acc[j][r] = e;
      rmax[r] = fmaxf(rmax[r], e);
    }
  }
#pragma unroll
  for (int s = 1; s < 16; s <<= 1)          // row lives in one 16-lane half
#pragma unroll
    for (int r = 0; r < 8; ++r) rmax[r] = fmaxf(rmax[r], __shfl_xor(rmax[r], s, 32));
#pragma unroll
  for (int j = 0; j < 6; ++j)
#pragma unroll
    for (int r = 0; r < 8; ++r) {
      float e = __expf(acc[j][r] - rmax[r]);
      acc[j][r] = e;
      rsum[r] += e;
    }
#pragma unroll
  for (int s = 1; s < 16; s <<= 1)
#pragma unroll
    for (int r = 0; r < 8; ++r) rsum[r] += __shfl_xor(rsum[r], s, 32);
#pragma unroll
  for (int j = 0; j < 6; ++j)
#pragma unroll
    for (int r = 0; r < 8; ++r)
      attnLDS[w][g * 8 + r][j * 16 + nl] = (__bf16)(acc[j][r] / rsum[r]);
  __syncthreads();

  // --- Y1 = attn @ V (12 WMMAs) --------------------------------------------
  v8f yacc[4];
#pragma unroll
  for (int j = 0; j < 4; ++j) yacc[j] = azero();
#pragma unroll
  for (int kc = 0; kc < 3; ++kc) {                     // K = 96 in steps of 32
    v16bf af = frag_ld_bf16(&attnLDS[w][nl][kc * 32 + g * 8]);
#pragma unroll
    for (int j = 0; j < 4; ++j) {                      // dh tiles
      v16bf bf = frag_ld_bf16(&vtLDS[j * 16 + nl][kc * 32 + g * 8]);
      yacc[j] = wmma_bf16(af, bf, yacc[j]);
    }
  }
#pragma unroll
  for (int j = 0; j < 4; ++j)
#pragma unroll
    for (int r = 0; r < 8; ++r) {
      int t = w * 16 + g * 8 + r;
      if (t < T_)
        y1[((size_t)a * T_ + t) * D_ + h * 64 + j * 16 + nl] = (__bf16)yacc[j][r];
    }
}

// ---------------------------------------------------------------------------
// K3: S = relu(Y1 @ W_Y2 + b) + x, stored bf16.  grid = MT, block = 128.
// Y1 tile staged to LDS once (was read 4x, once per wave).
// ---------------------------------------------------------------------------
__global__ __launch_bounds__(128)
void y2_residual(const __bf16* __restrict__ y1, const __bf16* __restrict__ wtY2,
                 const float* __restrict__ bY2, const float* __restrict__ x,
                 __bf16* __restrict__ sOut) {
  const int rt = blockIdx.x;
  const int tid = threadIdx.x, lane = tid & 31, w = tid >> 5;
  const int nl = lane & 15, g = lane >> 4;

  __shared__ __bf16 yTile[16][256];        // 8 KB
  for (int i = tid; i < 16 * 256; i += 128) {
    int r0 = i >> 8, c = i & 255;
    int rr = rt * 16 + r0;
    yTile[r0][c] = (rr < MROWS) ? y1[(size_t)rr * D_ + c] : (__bf16)0.0f;
  }
  __syncthreads();

  v16bf af[8];
#pragma unroll
  for (int c = 0; c < 8; ++c) af[c] = frag_ld_bf16(&yTile[nl][c * 32 + g * 8]);

  v8f acc[4];
#pragma unroll
  for (int j = 0; j < 4; ++j) acc[j] = azero();
#pragma unroll
  for (int kc = 0; kc < 8; ++kc) {
#pragma unroll
    for (int j = 0; j < 4; ++j) {
      int col = w * 64 + j * 16 + nl;
      v16bf bf = frag_ld_bf16(wtY2 + (size_t)col * D_ + kc * 32 + g * 8);
      acc[j] = wmma_bf16(af[kc], bf, acc[j]);
    }
  }
#pragma unroll
  for (int j = 0; j < 4; ++j) {
    int   col = w * 64 + j * 16 + nl;
    float bv  = bY2[col];
#pragma unroll
    for (int r = 0; r < 8; ++r) {
      int rr = rt * 16 + g * 8 + r;
      if (rr < MROWS) {
        float v = acc[j][r] + bv;
        v = v > 0.0f ? v : 0.0f;
        v += x[(size_t)rr * D_ + col];                 // residual (fp32)
        sOut[(size_t)rr * D_ + col] = (__bf16)v;
      }
    }
  }
}

// ---------------------------------------------------------------------------
// K4: fused FFN + LayerNorm.  grid = MT, block = 256 (8 waves).
// S tile (one contiguous 8 KB block) arrives via the Tensor Data Mover;
// F1 (16x1024) lives in LDS as bf16; F2 (16x256) in LDS as fp32; then LN -> Z.
// ---------------------------------------------------------------------------
__global__ __launch_bounds__(256)
void ffn_ln(const __bf16* __restrict__ s, const __bf16* __restrict__ wtF1,
            const float* __restrict__ bF1, const __bf16* __restrict__ wtF2,
            const float* __restrict__ bF2, const float* __restrict__ ln_g,
            const float* __restrict__ ln_b, float* __restrict__ z) {
  const int rt = blockIdx.x;
  const int tid = threadIdx.x, lane = tid & 31, w = tid >> 5;
  const int nl = lane & 15, g = lane >> 4;

  __shared__ __bf16 sTile[16][256];    //  8 KB
  __shared__ __bf16 f1Tile[16][1024];  // 32 KB
  __shared__ float  f2Tile[16][256];   // 16 KB

  // TDM: one wave issues the async DMA of the whole (contiguous) S tile.
  if (w == 0) {
    unsigned lds_off = (unsigned)(unsigned long long)(&sTile[0][0]);
    tdm_load_row_bf16(lds_off, s + (size_t)rt * 16 * D_, 16 * D_);
    __builtin_amdgcn_s_wait_tensorcnt(0);
  }
  __syncthreads();

  // --- F1 = relu(S @ W_F1 + b): 8 column tiles per wave, K = 256 -----------
  v16bf af[8];
#pragma unroll
  for (int c = 0; c < 8; ++c) af[c] = frag_ld_bf16(&sTile[nl][c * 32 + g * 8]);

#pragma unroll
  for (int j = 0; j < 8; ++j) {
    int col = w * 128 + j * 16 + nl;
    v8f acc = azero();
#pragma unroll
    for (int c = 0; c < 8; ++c) {
      v16bf bf = frag_ld_bf16(wtF1 + (size_t)col * D_ + c * 32 + g * 8);
      acc = wmma_bf16(af[c], bf, acc);
    }
    float bv = bF1[col];
#pragma unroll
    for (int r = 0; r < 8; ++r) {
      float v = acc[r] + bv;
      f1Tile[g * 8 + r][col] = (__bf16)(v > 0.0f ? v : 0.0f);
    }
  }
  __syncthreads();

  // --- F2 = relu(F1 @ W_F2 + b): 2 column tiles per wave, K = 1024 ---------
#pragma unroll
  for (int j = 0; j < 2; ++j) {
    int col = w * 32 + j * 16 + nl;
    __builtin_prefetch(wtF2 + (size_t)col * KD_, 0, 0);   // global_prefetch_b8
    v8f acc = azero();
    for (int c = 0; c < 32; ++c) {
      v16bf a2 = frag_ld_bf16(&f1Tile[nl][c * 32 + g * 8]);
      v16bf b2 = frag_ld_bf16(wtF2 + (size_t)col * KD_ + c * 32 + g * 8);
      acc = wmma_bf16(a2, b2, acc);
    }
    float bv = bF2[col];
#pragma unroll
    for (int r = 0; r < 8; ++r) {
      float v = acc[r] + bv;
      f2Tile[g * 8 + r][col] = v > 0.0f ? v : 0.0f;
    }
  }
  __syncthreads();

  // --- LayerNorm over D=256, one row per thread ----------------------------
  if (tid < 16) {
    int rr = rt * 16 + tid;
    if (rr < MROWS) {
      float mu = 0.0f;
      for (int c = 0; c < 256; ++c) mu += f2Tile[tid][c];
      mu *= (1.0f / 256.0f);
      float var = 0.0f;
      for (int c = 0; c < 256; ++c) { float d0 = f2Tile[tid][c] - mu; var += d0 * d0; }
      var *= (1.0f / 256.0f);
      float inv = rsqrtf(var + 1e-5f);
      for (int c = 0; c < 256; ++c)
        z[(size_t)rr * D_ + c] = (f2Tile[tid][c] - mu) * inv * ln_g[c] + ln_b[c];
    }
  }
}

// ---------------------------------------------------------------------------
extern "C" void kernel_launch(void* const* d_in, const int* in_sizes, int n_in,
                              void* d_out, int out_size, void* d_ws, size_t ws_size,
                              hipStream_t stream) {
  (void)in_sizes; (void)n_in; (void)out_size; (void)ws_size;
  const float* x   = (const float*)d_in[0];
  const unsigned char* pad = (const unsigned char*)d_in[2];   // bool mask
  const float* W_K  = (const float*)d_in[3];  const float* b_K  = (const float*)d_in[4];
  const float* W_V  = (const float*)d_in[5];  const float* b_V  = (const float*)d_in[6];
  const float* W_Q  = (const float*)d_in[7];  const float* b_Q  = (const float*)d_in[8];
  const float* scale_q = (const float*)d_in[9];
  const float* W_Y2 = (const float*)d_in[10]; const float* b_Y2 = (const float*)d_in[11];
  const float* W_F1 = (const float*)d_in[12]; const float* b_F1 = (const float*)d_in[13];
  const float* W_F2 = (const float*)d_in[14]; const float* b_F2 = (const float*)d_in[15];
  const float* ln_g = (const float*)d_in[16]; const float* ln_b = (const float*)d_in[17];

  char* ws = (char*)d_ws;
  __bf16* wtK  = (__bf16*)(ws);
  __bf16* wtV  = (__bf16*)(ws + 131072);
  __bf16* wtQ  = (__bf16*)(ws + 262144);
  __bf16* wtY2 = (__bf16*)(ws + 393216);
  __bf16* wtF1 = (__bf16*)(ws + 524288);    // [1024][256]
  __bf16* wtF2 = (__bf16*)(ws + 1048576);   // [256][1024]
  __bf16* y1   = (__bf16*)(ws + 1572864);                                  // [MROWS][256]
  __bf16* sbuf = (__bf16*)(ws + 1572864 + (size_t)MROWS * D_ * 2);         // [MROWS][256]

  float* out = (float*)d_out;   // Z | Qp | Kp | Vp

  // K0: bf16 transposed weights
  transpose_bf16<<<dim3(256),  256, 0, stream>>>(wtK,  W_K,  256,  256);
  transpose_bf16<<<dim3(256),  256, 0, stream>>>(wtV,  W_V,  256,  256);
  transpose_bf16<<<dim3(256),  256, 0, stream>>>(wtQ,  W_Q,  256,  256);
  transpose_bf16<<<dim3(256),  256, 0, stream>>>(wtY2, W_Y2, 256,  256);
  transpose_bf16<<<dim3(1024), 256, 0, stream>>>(wtF1, W_F1, 256,  1024);
  transpose_bf16<<<dim3(1024), 256, 0, stream>>>(wtF2, W_F2, 1024, 256);

  // K1..K4
  qkv_proj   <<<dim3(MT),     384, 0, stream>>>(x, wtK, wtV, wtQ, b_K, b_V, b_Q, scale_q, out);
  attn_kernel<<<dim3(A_, H_), 192, 0, stream>>>(out, pad, y1);
  y2_residual<<<dim3(MT),     128, 0, stream>>>(y1, wtY2, b_Y2, x, sbuf);
  ffn_ln     <<<dim3(MT),     256, 0, stream>>>(sbuf, wtF1, b_F1, wtF2, b_F2, ln_g, ln_b, out);
}
